// SchNetLayer_53257594471008
// MI455X (gfx1250) — matrix operand order repaired and checked by reading, not compiled, not run on hardware
//
#include <hip/hip_runtime.h>

// ---------------------------------------------------------------------------
// SchNet layer for MI455X (gfx1250): bf16 WMMA for all GEMMs, native f32
// global atomics for the edge scatter-add, L2-resident x1/agg arrays.
// ---------------------------------------------------------------------------

#define NNODE   100000
#define NEDGE   1600000
#define HDIM    128
#define FDIM    128
#define GDIM    51
#define GPAD    64
#define WAVES   8          // waves per workgroup (wave32)
#define LOG2F_C 0.69314718f

typedef __attribute__((ext_vector_type(16))) __bf16 v16bf;
typedef __attribute__((ext_vector_type(8)))  float  v8f;

union V16U { v16bf v; uint4 q[2]; };

// A-operand (16-bit 16xK tile, row-major in LDS with stride AS elements).
// ISA layout: lane L holds row M=L&15; K halves at ((L&16)?8:0) and +16.
template<int AS>
__device__ inline v16bf load_A(const __bf16* At, int lane, int ks) {
  int m  = lane & 15;
  int k0 = ks * 32 + ((lane & 16) >> 1);     // 0 or 8
  const __bf16* p = At + m * AS + k0;
  V16U u;
  u.q[0] = *(const uint4*)(p);               // K = k0 .. k0+7
  u.q[1] = *(const uint4*)(p + 16);          // K = k0+16 .. k0+23
  return u.v;
}

// B-operand from bf16 weights stored [outdim][BKP] row-major (== B^T), so a
// lane reads K halves at ((L&16)?8:0) and +16 of column N = nt*16 + (lane&15).
template<int BKP>
__device__ inline v16bf load_B(const __bf16* __restrict__ Bw, int lane, int nt, int ks) {
  int n  = nt * 16 + (lane & 15);
  int k0 = ks * 32 + ((lane & 16) >> 1);     // 0 or 8
  const __bf16* p = Bw + n * BKP + k0;
  V16U u;
  u.q[0] = *(const uint4*)(p);               // K = k0 .. k0+7
  u.q[1] = *(const uint4*)(p + 16);          // K = k0+16 .. k0+23
  return u.v;
}

// One wave computes a full 16 x 128 output tile: 8 N-tiles, NK k-steps of 32.
template<int NK, int AS, int BKP>
__device__ inline void wave_gemm(const __bf16* At, const __bf16* __restrict__ Bw,
                                 int lane, v8f* acc) {
  v8f zv = {};
#pragma unroll
  for (int nt = 0; nt < 8; ++nt) acc[nt] = zv;
#pragma unroll
  for (int ks = 0; ks < NK; ++ks) {
    v16bf a = load_A<AS>(At, lane, ks);
#pragma unroll
    for (int nt = 0; nt < 8; ++nt) {
      v16bf b = load_B<BKP>(Bw, lane, nt, ks);
      acc[nt] = __builtin_amdgcn_wmma_f32_16x16x32_bf16(
          false, a, false, b, (short)0, acc[nt], false, false);
    }
  }
}

__device__ inline float ssp_f(float x) {
  // softplus(x) - log(2), numerically stable
  return fmaxf(x, 0.0f) + __logf(1.0f + __expf(-fabsf(x))) - LOG2F_C;
}

// --------------------------- weight prep -----------------------------------
__global__ void prep_kernel(const float* __restrict__ mlp0_w, const float* __restrict__ mlp2_w,
                            const float* __restrict__ lin1_w, const float* __restrict__ lin2_w,
                            const float* __restrict__ blk_w,  const float* __restrict__ out1_w,
                            const float* __restrict__ out2_w,
                            __bf16* w0b, __bf16* w2b, __bf16* l1b, __bf16* l2b,
                            __bf16* bkb, __bf16* o1b, __bf16* o2b) {
  int i = blockIdx.x * blockDim.x + threadIdx.x;
  if (i < FDIM * GPAD) {
    int r = i >> 6, c = i & 63;
    w0b[i] = (__bf16)((c < GDIM) ? mlp0_w[r * GDIM + c] : 0.0f);
  }
  if (i < FDIM * FDIM) {
    w2b[i] = (__bf16)mlp2_w[i];
    l1b[i] = (__bf16)lin1_w[i];
    l2b[i] = (__bf16)lin2_w[i];
    bkb[i] = (__bf16)blk_w[i];
    o1b[i] = (__bf16)out1_w[i];
    o2b[i] = (__bf16)out2_w[i];
  }
}

// --------------------------- x1 = z @ lin1^T -------------------------------
__global__ __launch_bounds__(WAVES * 32) void lin1_kernel(
    const float* __restrict__ z, const __bf16* __restrict__ l1b, float* __restrict__ x1) {
  __shared__ __align__(16) __bf16 zt[WAVES][16 * 136];
  int wid = threadIdx.x >> 5, lane = threadIdx.x & 31;
  int tile = blockIdx.x * WAVES + wid;
  if (tile >= NNODE / 16) return;
  int node0 = tile * 16;
  int m = lane & 15;
  int c0 = (lane & 16) << 2;                 // 0 or 64
  const float* zp = z + (long long)(node0 + m) * HDIM + c0;
#pragma unroll 8
  for (int j = 0; j < 64; ++j) zt[wid][m * 136 + c0 + j] = (__bf16)zp[j];

  v8f acc[8];
  wave_gemm<4, 136, 128>(zt[wid], l1b, lane, acc);

  int n0 = lane & 15, mh = (lane & 16) >> 1;
#pragma unroll
  for (int nt = 0; nt < 8; ++nt)
#pragma unroll
    for (int r = 0; r < 8; ++r)
      x1[(long long)(node0 + r + mh) * FDIM + nt * 16 + n0] = acc[nt][r];
}

// ---------------- edge filter + gather*W + scatter-add ----------------------
__global__ __launch_bounds__(WAVES * 32) void edge_kernel(
    const long long* __restrict__ ei, const float* __restrict__ pos,
    const float* __restrict__ x1,
    const __bf16* __restrict__ w0b, const float* __restrict__ mlp0_b,
    const __bf16* __restrict__ w2b, const float* __restrict__ mlp2_b,
    float* __restrict__ agg) {
  __shared__ __align__(16) __bf16 eaS[WAVES][16 * 72];
  __shared__ __align__(16) __bf16 h1S[WAVES][16 * 136];
  __shared__ int   rowS[WAVES][16];
  __shared__ int   colS[WAVES][16];
  __shared__ float cS[WAVES][16];

  int wid = threadIdx.x >> 5, lane = threadIdx.x & 31;
  int tile = blockIdx.x * WAVES + wid;       // NEDGE/16 == 100000 == grid*WAVES
  int e0 = tile * 16;
  int m = lane & 15;

  // per-edge geometry (lanes 16-31 redundantly compute their m's edge)
  int e  = e0 + m;
  int rw = (int)ei[e];
  int cl = (int)ei[NEDGE + e];
  float dx = pos[3 * rw + 0] - pos[3 * cl + 0];
  float dy = pos[3 * rw + 1] - pos[3 * cl + 1];
  float dz = pos[3 * rw + 2] - pos[3 * cl + 2];
  float ew = sqrtf(dx * dx + dy * dy + dz * dz + 1e-12f);
  if (lane < 16) {
    rowS[wid][m] = rw;
    colS[wid][m] = cl;
    cS[wid][m]   = 0.5f * (__cosf(ew * 0.31415927f) + 1.0f);   // pi/cutoff
  }
  // Gaussian smearing: offset g*0.2, coeff -12.5; pad 51->64 with zeros
  int gbase = (lane & 16) << 1;              // 0 or 32
#pragma unroll 8
  for (int j = 0; j < 32; ++j) {
    int g = gbase + j;
    float t = ew - 0.2f * (float)g;
    float v = (g < GDIM) ? __expf(-12.5f * t * t) : 0.0f;
    eaS[wid][m * 72 + g] = (__bf16)v;
  }

  v8f acc[8];
  int n0 = lane & 15, mh = (lane & 16) >> 1;

  // GEMM1: [16x64] x [64x128] -> ssp -> bf16 tile
  wave_gemm<2, 72, 64>(eaS[wid], w0b, lane, acc);
#pragma unroll
  for (int nt = 0; nt < 8; ++nt) {
    float bias = mlp0_b[nt * 16 + n0];
#pragma unroll
    for (int r = 0; r < 8; ++r)
      h1S[wid][(r + mh) * 136 + nt * 16 + n0] = (__bf16)ssp_f(acc[nt][r] + bias);
  }

  // GEMM2: [16x128] x [128x128] -> * cutoff -> W; msg = x1[row]*W -> agg[col]
  wave_gemm<4, 136, 128>(h1S[wid], w2b, lane, acc);
#pragma unroll
  for (int r = 0; r < 8; ++r) {
    int mm = r + mh;
    int rr = rowS[wid][mm];
    int cc = colS[wid][mm];
    float cf = cS[wid][mm];
#pragma unroll
    for (int nt = 0; nt < 8; ++nt) {
      int n = nt * 16 + n0;
      float wv = (acc[nt][r] + mlp2_b[n]) * cf;
      float xv = x1[(long long)rr * FDIM + n];
      unsafeAtomicAdd(&agg[(long long)cc * FDIM + n], xv * wv);
    }
  }
}

// -------- conv=agg@lin2+b; h=z+ssp(conv)@blk+b; out=ssp(h@o1+b)@o2+b --------
__global__ __launch_bounds__(WAVES * 32) void post_kernel(
    const float* __restrict__ z, const float* __restrict__ agg_in,
    const __bf16* __restrict__ l2b, const float* __restrict__ l2bias,
    const __bf16* __restrict__ bkb, const float* __restrict__ bkbias,
    const __bf16* __restrict__ o1b, const float* __restrict__ o1bias,
    const __bf16* __restrict__ o2b, const float* __restrict__ o2bias,
    float* __restrict__ out) {
  __shared__ __align__(16) __bf16 tA[WAVES][16 * 136];
  __shared__ __align__(16) __bf16 tB[WAVES][16 * 136];
  int wid = threadIdx.x >> 5, lane = threadIdx.x & 31;
  int tile = blockIdx.x * WAVES + wid;
  if (tile >= NNODE / 16) return;
  int node0 = tile * 16;
  int m = lane & 15;
  int c0 = (lane & 16) << 2;
  int n0 = lane & 15, mh = (lane & 16) >> 1;

  // stage agg tile (read BEFORE we overwrite the same region of d_out)
  const float* ap = agg_in + (long long)(node0 + m) * FDIM + c0;
#pragma unroll 8
  for (int j = 0; j < 64; ++j) tA[wid][m * 136 + c0 + j] = (__bf16)ap[j];

  v8f acc[8];

  // conv = agg @ lin2^T + b  ->  ssp  ->  tB
  wave_gemm<4, 136, 128>(tA[wid], l2b, lane, acc);
#pragma unroll
  for (int nt = 0; nt < 8; ++nt) {
    float bias = l2bias[nt * 16 + n0];
#pragma unroll
    for (int r = 0; r < 8; ++r)
      tB[wid][(r + mh) * 136 + nt * 16 + n0] = (__bf16)ssp_f(acc[nt][r] + bias);
  }

  // h = z + tB @ blk^T + b  ->  tA
  wave_gemm<4, 136, 128>(tB[wid], bkb, lane, acc);
#pragma unroll
  for (int nt = 0; nt < 8; ++nt) {
    float bias = bkbias[nt * 16 + n0];
#pragma unroll
    for (int r = 0; r < 8; ++r) {
      float h = z[(long long)(node0 + r + mh) * HDIM + nt * 16 + n0] + acc[nt][r] + bias;
      tA[wid][(r + mh) * 136 + nt * 16 + n0] = (__bf16)h;
    }
  }

  // ssp(h @ out1^T + b)  ->  tB
  wave_gemm<4, 136, 128>(tA[wid], o1b, lane, acc);
#pragma unroll
  for (int nt = 0; nt < 8; ++nt) {
    float bias = o1bias[nt * 16 + n0];
#pragma unroll
    for (int r = 0; r < 8; ++r)
      tB[wid][(r + mh) * 136 + nt * 16 + n0] = (__bf16)ssp_f(acc[nt][r] + bias);
  }

  // out = tB @ out2^T + b
  wave_gemm<4, 136, 128>(tB[wid], o2b, lane, acc);
#pragma unroll
  for (int nt = 0; nt < 8; ++nt) {
    float bias = o2bias[nt * 16 + n0];
#pragma unroll
    for (int r = 0; r < 8; ++r)
      out[(long long)(node0 + r + mh) * HDIM + nt * 16 + n0] = acc[nt][r] + bias;
  }
}

// ---------------------------------------------------------------------------
extern "C" void kernel_launch(void* const* d_in, const int* in_sizes, int n_in,
                              void* d_out, int out_size, void* d_ws, size_t ws_size,
                              hipStream_t stream) {
  const float*     z      = (const float*)d_in[0];
  const float*     pos    = (const float*)d_in[1];
  const long long* ei     = (const long long*)d_in[2];
  const float*     lin1_w = (const float*)d_in[3];
  const float*     lin2_w = (const float*)d_in[4];
  const float*     lin2_b = (const float*)d_in[5];
  const float*     mlp0_w = (const float*)d_in[6];
  const float*     mlp0_b = (const float*)d_in[7];
  const float*     mlp2_w = (const float*)d_in[8];
  const float*     mlp2_b = (const float*)d_in[9];
  const float*     blk_w  = (const float*)d_in[10];
  const float*     blk_b  = (const float*)d_in[11];
  const float*     out1_w = (const float*)d_in[12];
  const float*     out1_b = (const float*)d_in[13];
  const float*     out2_w = (const float*)d_in[14];
  const float*     out2_b = (const float*)d_in[15];

  // workspace: bf16 weight copies, then x1 [N,128] f32; agg aliases d_out
  __bf16* w0b = (__bf16*)d_ws;                 // 128*64
  __bf16* w2b = w0b + FDIM * GPAD;             // 128*128
  __bf16* l1b = w2b + FDIM * FDIM;
  __bf16* l2b = l1b + FDIM * FDIM;
  __bf16* bkb = l2b + FDIM * FDIM;
  __bf16* o1b = bkb + FDIM * FDIM;
  __bf16* o2b = o1b + FDIM * FDIM;
  float*  x1  = (float*)(o2b + FDIM * FDIM);   // byte offset 212992 (16B aligned)
  float*  agg = (float*)d_out;

  hipMemsetAsync(d_out, 0, (size_t)out_size * sizeof(float), stream);

  prep_kernel<<<(FDIM * FDIM + 255) / 256, 256, 0, stream>>>(
      mlp0_w, mlp2_w, lin1_w, lin2_w, blk_w, out1_w, out2_w,
      w0b, w2b, l1b, l2b, bkb, o1b, o2b);

  int ntile_n = NNODE / 16;                        // 6250
  lin1_kernel<<<(ntile_n + WAVES - 1) / WAVES, WAVES * 32, 0, stream>>>(z, l1b, x1);

  int ntile_e = NEDGE / 16;                        // 100000, divisible by WAVES
  edge_kernel<<<ntile_e / WAVES, WAVES * 32, 0, stream>>>(
      ei, pos, x1, w0b, mlp0_b, w2b, mlp2_b, agg);

  post_kernel<<<(ntile_n + WAVES - 1) / WAVES, WAVES * 32, 0, stream>>>(
      z, agg, l2b, lin2_b, bkb, blk_b, o1b, out1_b, o2b, out2_b, (float*)d_out);
}
